// UncertaintyWeightedCrossAttention_74594991997563
// MI455X (gfx1250) — matrix operand order repaired and checked by reading, not compiled, not run on hardware
//
#include <hip/hip_runtime.h>
#include <cstdint>
#include <cstddef>

// Problem dims (fixed by reference)
#define BB   8
#define MM   1024
#define HH   1024
#define NH   16
#define HD   64
#define ROWS (BB * MM)   // 8192

typedef __attribute__((ext_vector_type(16))) __bf16       bf16x16;
typedef __attribute__((ext_vector_type(8)))  float        f32x8;
typedef __attribute__((ext_vector_type(4)))  unsigned int u32x4;

union BFrag {
  bf16x16 v;
  u32x4 q[2];
  unsigned short us[16];
};

__device__ __forceinline__ unsigned short f2bf(float f) {
  union { float f; unsigned int u; } v; v.f = f;
  unsigned int u = v.u;
  return (unsigned short)((u + 0x7FFFu + ((u >> 16) & 1u)) >> 16); // RNE
}

__device__ __forceinline__ f32x8 wmma_bf16(bf16x16 a, bf16x16 b, f32x8 c) {
  // 8 args: (neg_a, A, neg_b, B, c_mod, C, reuse_a, reuse_b)
  return __builtin_amdgcn_wmma_f32_16x16x32_bf16(false, a, false, b, (short)0, c,
                                                 false, false);
}

// A-matrix 16x32 bf16 fragment, row-major source (stride ld), ISA §7.12.2 layout:
// lanes 0-15: row M=lane, K 0..7 (v0..3) and 16..23 (v4..7); lanes 16-31: K 8..15 / 24..31
__device__ __forceinline__ bf16x16 load_a_frag(const unsigned short* base, int ld, int lane) {
  const int r = lane & 15, half = lane >> 4;
  const unsigned short* p = base + (size_t)r * ld + half * 8;
  BFrag f;
  f.q[0] = *reinterpret_cast<const u32x4*>(p);
  f.q[1] = *reinterpret_cast<const u32x4*>(p + 16);
  return f.v;
}

// B-matrix 32x16 bf16 fragment, stored N-major (row = N column, contiguous along K):
// lanes 0-15: N=lane, K 0..7 and 16..23; lanes 16-31: N=lane-16, K 8..15 / 24..31
__device__ __forceinline__ bf16x16 load_b_frag(const unsigned short* base, int ld, int lane) {
  const int n = lane & 15, half = lane >> 4;
  const unsigned short* p = base + (size_t)n * ld + half * 8;
  BFrag f;
  f.q[0] = *reinterpret_cast<const u32x4*>(p);
  f.q[1] = *reinterpret_cast<const u32x4*>(p + 16);
  return f.v;
}

// A fragment assembled from an f32 LDS strip (16 rows x 1024), converting to bf16
__device__ __forceinline__ bf16x16 a_frag_from_lds(const float* s, int k0, int lane) {
  const int r = lane & 15, half = lane >> 4;
  const float* p = s + r * 1024 + k0 + half * 8;
  BFrag f;
#pragma unroll
  for (int i = 0; i < 8; ++i) f.us[i] = f2bf(p[i]);
#pragma unroll
  for (int i = 0; i < 8; ++i) f.us[8 + i] = f2bf(p[16 + i]);
  return f.v;
}

// ---------------------------------------------------------------- conversions
__global__ void cvt_f32_bf16_kernel(const float* __restrict__ in,
                                    unsigned short* __restrict__ out, int n) {
  int i = blockIdx.x * 256 + threadIdx.x;
  if (i < n) out[i] = f2bf(in[i]);
}

// WT[n][k] = bf16(W[k][n]); W is 1024x1024 f32
__global__ void transpose_w_kernel(const float* __restrict__ W,
                                   unsigned short* __restrict__ WT) {
  __shared__ float tile[32][33];
  const int k0 = blockIdx.x * 32, n0 = blockIdx.y * 32;
  const int tx = threadIdx.x & 31, ty = threadIdx.x >> 5; // 8 row-groups
  for (int i = ty; i < 32; i += 8) tile[i][tx] = W[(size_t)(k0 + i) * HH + n0 + tx];
  __syncthreads();
  for (int i = ty; i < 32; i += 8) WT[(size_t)(n0 + i) * HH + k0 + tx] = f2bf(tile[tx][i]);
}

__global__ void zero_f32_kernel(float* p, int n) {
  int i = blockIdx.x * 256 + threadIdx.x;
  if (i < n) p[i] = 0.0f;
}

// ---------------------------------------------------------------- projections
// Y = X(bf16, ROWSx1024) @ W + b, using WT (N-major bf16).
// Each wave computes a 16x64 strip: one A-fragment feeds 4 WMMA accumulators.
// mode 0: f32 out [ROWS,1024], sigmoid (gate)
// mode 1: bf16 out [ROWS,1024]          (q, k: [b,m,h,d] contiguous in d)
// mode 2: bf16 out head-transposed vT[((b*16+h)*64+d)*1024 + m]
__device__ __forceinline__ void store_proj_tile(const f32x8& acc, int m0, int col0,
                                                const float* bias, void* out, int mode,
                                                int lane) {
  const int n  = lane & 15;
  const int mb = (lane >> 4) * 8;
  const int col = col0 + n;
  const float bv = bias[col];
#pragma unroll
  for (int i = 0; i < 8; ++i) {
    const int row = m0 + mb + i;
    const float y = acc[i] + bv;
    if (mode == 0) {
      ((float*)out)[(size_t)row * HH + col] = 1.0f / (1.0f + __expf(-y));
    } else if (mode == 1) {
      ((unsigned short*)out)[(size_t)row * HH + col] = f2bf(y);
    } else {
      const int b_ = row >> 10, m_ = row & 1023, h_ = col >> 6, d_ = col & 63;
      ((unsigned short*)out)[(((size_t)(b_ * NH + h_) * HD + d_) << 10) + m_] = f2bf(y);
    }
  }
}

__global__ void proj_gemm_kernel(const unsigned short* __restrict__ X,
                                 const unsigned short* __restrict__ WT,
                                 const float* __restrict__ bias,
                                 void* __restrict__ out, int mode) {
  const int lane = threadIdx.x & 31;
  const int wid  = threadIdx.x >> 5;             // 0..7
  const int m0   = blockIdx.x * 16;              // 512 row tiles
  const int n0   = (blockIdx.y * 8 + wid) * 64;  // blockIdx.y 0..1 -> 16 waves cover 1024
  const unsigned short* Arow = X + (size_t)m0 * HH;
  const unsigned short* B0 = WT + (size_t)(n0 +  0) * HH;
  const unsigned short* B1 = WT + (size_t)(n0 + 16) * HH;
  const unsigned short* B2 = WT + (size_t)(n0 + 32) * HH;
  const unsigned short* B3 = WT + (size_t)(n0 + 48) * HH;
  f32x8 acc0 = {}, acc1 = {}, acc2 = {}, acc3 = {};
  for (int k0 = 0; k0 < HH; k0 += 32) {
    if (k0 + 32 < HH) {
      __builtin_prefetch(Arow + k0 + 32, 0, 1);
      __builtin_prefetch(B0 + k0 + 32, 0, 1);
    }
    const bf16x16 a = load_a_frag(Arow + k0, HH, lane);   // shared across 4 tiles
    acc0 = wmma_bf16(a, load_b_frag(B0 + k0, HH, lane), acc0);
    acc1 = wmma_bf16(a, load_b_frag(B1 + k0, HH, lane), acc1);
    acc2 = wmma_bf16(a, load_b_frag(B2 + k0, HH, lane), acc2);
    acc3 = wmma_bf16(a, load_b_frag(B3 + k0, HH, lane), acc3);
  }
  store_proj_tile(acc0, m0, n0 +  0, bias, out, mode, lane);
  store_proj_tile(acc1, m0, n0 + 16, bias, out, mode, lane);
  store_proj_tile(acc2, m0, n0 + 32, bias, out, mode, lane);
  store_proj_tile(acc3, m0, n0 + 48, bias, out, mode, lane);
}

// ---------------------------------------------------------------- sigma MLP
__global__ void sigma_kernel(const float* __restrict__ X,
                             const float* __restrict__ Wu1, const float* __restrict__ bu1,
                             const float* __restrict__ Wu2, const float* __restrict__ bu2,
                             float* __restrict__ sig_ws, float* __restrict__ sig_out) {
  __shared__ float xr[HH];
  __shared__ float hred[256];
  const int row = blockIdx.x;   // 0..8191
  const int t   = threadIdx.x;  // 0..255
  for (int i = t; i < HH; i += 256) xr[i] = X[(size_t)row * HH + i];
  __syncthreads();
  float acc = bu1[t];
  for (int k = 0; k < HH; ++k) acc += xr[k] * Wu1[(size_t)k * 256 + t];
  acc = fmaxf(acc, 0.0f);                 // ReLU
  hred[t] = acc * Wu2[t];
  __syncthreads();
  for (int s = 128; s > 0; s >>= 1) {
    if (t < s) hred[t] += hred[t + s];
    __syncthreads();
  }
  if (t == 0) {
    const float z  = hred[0] + bu2[0];
    const float sp = (z > 20.0f) ? z : log1pf(__expf(z));  // softplus
    const float sg = sp + 1e-6f;
    sig_ws[row]  = sg;
    sig_out[row] = sg;
  }
}

// ---------------------------------------------------------------- attention
// One workgroup per (b, head, 16-row m-tile). 8 waves.
__global__ void uwca_attn_kernel(const unsigned short* __restrict__ Q,
                                 const unsigned short* __restrict__ K,
                                 const unsigned short* __restrict__ VT,
                                 const float* __restrict__ G,
                                 const float* __restrict__ SIG,
                                 float* __restrict__ attn_out,
                                 float* __restrict__ Zsum) {
  __shared__ float sc[16 * 1024];      // 64 KB score strip
  __shared__ float red[16 * 16];
  __shared__ float rowstat[16];
  __shared__ float invsig_m[16];
  __shared__ float ctxpart[4][256];    // upper-K partial ctx accumulators
  const int lane = threadIdx.x & 31;
  const int wid  = threadIdx.x >> 5;
  const int m0   = blockIdx.x * 16;       // 64 m-tiles
  const int h    = blockIdx.y;            // 16 heads
  const int b    = blockIdx.z;            // 8 batch
  const int bh   = b * NH + h;

  if (threadIdx.x < 16) invsig_m[threadIdx.x] = 1.0f / SIG[b * MM + m0 + threadIdx.x];
  __syncthreads();

  // Q fragments for d-halves 0..31 and 32..63, kept in registers
  const unsigned short* qbase = Q + ((size_t)(b * MM + m0) * HH) + h * HD;
  const bf16x16 qlo = load_a_frag(qbase,      HH, lane);
  const bf16x16 qhi = load_a_frag(qbase + 32, HH, lane);

  const float rs = 0.125f; // 1/sqrt(64)
  for (int nt = wid; nt < 64; nt += 8) {
    const int n0 = nt * 16;
    const unsigned short* kbase = K + ((size_t)(b * MM + n0) * HH) + h * HD;
    const bf16x16 klo = load_b_frag(kbase,      HH, lane);
    const bf16x16 khi = load_b_frag(kbase + 32, HH, lane);
    f32x8 acc = {};
    acc = wmma_bf16(qlo, klo, acc);
    acc = wmma_bf16(qhi, khi, acc);
    const int n  = lane & 15;
    const int mb = (lane >> 4) * 8;
    const float isn = rs / SIG[b * MM + n0 + n];
#pragma unroll
    for (int i = 0; i < 8; ++i)
      sc[(mb + i) * 1024 + n0 + n] = acc[i] * isn * invsig_m[mb + i];
  }
  __syncthreads();

  // row softmax: 16 threads per row, 64 contiguous cols per thread
  const int r     = threadIdx.x & 15;
  const int chunk = threadIdx.x >> 4;
  float* srow = sc + r * 1024 + chunk * 64;
  float mx = -1e30f;
  for (int j = 0; j < 64; ++j) mx = fmaxf(mx, srow[j]);
  red[r * 16 + chunk] = mx;
  __syncthreads();
  if (chunk == 0) {
    float m = red[r * 16];
    for (int j = 1; j < 16; ++j) m = fmaxf(m, red[r * 16 + j]);
    rowstat[r] = m;
  }
  __syncthreads();
  const float rmax = rowstat[r];
  float ssum = 0.0f;
  for (int j = 0; j < 64; ++j) {
    const float e = __expf(srow[j] - rmax);
    srow[j] = e;
    ssum += e;
  }
  red[r * 16 + chunk] = ssum;
  __syncthreads();
  if (chunk == 0) {
    float t = 0.0f;
    for (int j = 0; j < 16; ++j) t += red[r * 16 + j];
    rowstat[r] = 1.0f / t;
  }
  __syncthreads();
  const float inv = rowstat[r];
  float* orow = attn_out + ((size_t)bh * MM + m0 + r) * MM + chunk * 64;
  for (int j = 0; j < 64; ++j) {
    const float a = srow[j] * inv;
    srow[j] = a;       // keep normalized attn in LDS for ctx WMMA
    orow[j] = a;       // stream 512MB attn output (HBM-bound term)
  }
  __syncthreads();

  // ctx = attn(16x1024) @ V(1024x64): all 8 waves.
  // wave -> (d-tile = wid&3, K-half = wid>>2); upper-K partials combined via LDS.
  {
    const int dt = wid & 3;
    const int kh = wid >> 2;
    const int d0 = dt * 16;
    const int kbeg = kh * 512, kend = kbeg + 512;
    const unsigned short* vbase = VT + (size_t)(bh * HD + d0) * MM;
    f32x8 acc = {};
    for (int k0 = kbeg; k0 < kend; k0 += 32) {
      bf16x16 a  = a_frag_from_lds(sc, k0, lane);
      bf16x16 bm = load_b_frag(vbase + k0, MM, lane);
      acc = wmma_bf16(a, bm, acc);
    }
    if (wid >= 4) {
#pragma unroll
      for (int i = 0; i < 8; ++i) ctxpart[dt][lane * 8 + i] = acc[i];
    }
    __syncthreads();
    if (wid < 4) {
      const int n  = lane & 15;
      const int mb = (lane >> 4) * 8;
      const int d  = d0 + n;
      float zacc = 0.0f;
#pragma unroll
      for (int i = 0; i < 8; ++i) {
        const float c = acc[i] + ctxpart[dt][lane * 8 + i];
        const int m = m0 + mb + i;
        const float gv = G[((size_t)(b * MM + m)) * HH + h * HD + d];
        zacc += c * gv;   // gated ctx, summed over this tile's m rows
      }
      atomicAdd(&Zsum[b * HH + h * HD + d], zacc);
    }
  }
}

// ---------------------------------------------------------------- output head
__global__ void finalize_kernel(const float* __restrict__ Zsum,
                                const float* __restrict__ Wo,
                                const float* __restrict__ bo,
                                float* __restrict__ z) {
  __shared__ float zr[HH];
  const int b = blockIdx.x;
  const int c = threadIdx.x;  // 1024 threads
  zr[c] = Zsum[b * HH + c] * (1.0f / (float)MM);  // mean over m
  __syncthreads();
  float acc = bo[c];
  for (int k = 0; k < HH; ++k) acc += zr[k] * Wo[(size_t)k * HH + c];
  z[b * HH + c] = acc;
}

// ---------------------------------------------------------------- launch
extern "C" void kernel_launch(void* const* d_in, const int* in_sizes, int n_in,
                              void* d_out, int out_size, void* d_ws, size_t ws_size,
                              hipStream_t stream) {
  (void)in_sizes; (void)n_in; (void)out_size; (void)ws_size;
  const float* x   = (const float*)d_in[0];
  const float* Wq  = (const float*)d_in[1];
  const float* bq  = (const float*)d_in[2];
  const float* Wk  = (const float*)d_in[3];
  const float* bk  = (const float*)d_in[4];
  const float* Wv  = (const float*)d_in[5];
  const float* bv  = (const float*)d_in[6];
  const float* Wg  = (const float*)d_in[7];
  const float* bg  = (const float*)d_in[8];
  const float* Wo  = (const float*)d_in[9];
  const float* bo  = (const float*)d_in[10];
  const float* Wu1 = (const float*)d_in[11];
  const float* bu1 = (const float*)d_in[12];
  const float* Wu2 = (const float*)d_in[13];
  const float* bu2 = (const float*)d_in[14];

  float* out_z     = (float*)d_out;                            // [8,1024]
  float* out_attn  = out_z + (size_t)BB * HH;                  // [8,16,1024,1024]
  float* out_sigma = out_attn + (size_t)BB * NH * MM * MM;     // [8,1024]

  char* ws = (char*)d_ws;
  auto take = [&](size_t bytes) { char* p = ws; ws += bytes; return p; };
  unsigned short* xbf = (unsigned short*)take((size_t)ROWS * HH * 2);  // 16 MB
  unsigned short* WTq = (unsigned short*)take((size_t)HH * HH * 2);    //  2 MB
  unsigned short* WTk = (unsigned short*)take((size_t)HH * HH * 2);
  unsigned short* WTv = (unsigned short*)take((size_t)HH * HH * 2);
  unsigned short* WTg = (unsigned short*)take((size_t)HH * HH * 2);
  unsigned short* qbf = (unsigned short*)take((size_t)ROWS * HH * 2);  // 16 MB
  unsigned short* kbf = (unsigned short*)take((size_t)ROWS * HH * 2);
  unsigned short* vT  = (unsigned short*)take((size_t)ROWS * HH * 2);
  float*          gbf = (float*)take((size_t)ROWS * HH * 4);           // 32 MB
  float*          sig = (float*)take((size_t)ROWS * 4);
  float*          Zs  = (float*)take((size_t)BB * HH * 4);

  // 1) precision conversion + weight transposes
  cvt_f32_bf16_kernel<<<dim3((ROWS * HH) / 256), 256, 0, stream>>>(x, xbf, ROWS * HH);
  transpose_w_kernel<<<dim3(32, 32), 256, 0, stream>>>(Wq, WTq);
  transpose_w_kernel<<<dim3(32, 32), 256, 0, stream>>>(Wk, WTk);
  transpose_w_kernel<<<dim3(32, 32), 256, 0, stream>>>(Wv, WTv);
  transpose_w_kernel<<<dim3(32, 32), 256, 0, stream>>>(Wg, WTg);

  // 2) WMMA projections (16x64 strip per wave, A-fragment reuse)
  proj_gemm_kernel<<<dim3(ROWS / 16, 2), 256, 0, stream>>>(xbf, WTq, bq, (void*)qbf, 1);
  proj_gemm_kernel<<<dim3(ROWS / 16, 2), 256, 0, stream>>>(xbf, WTk, bk, (void*)kbf, 1);
  proj_gemm_kernel<<<dim3(ROWS / 16, 2), 256, 0, stream>>>(xbf, WTv, bv, (void*)vT, 2);
  proj_gemm_kernel<<<dim3(ROWS / 16, 2), 256, 0, stream>>>(xbf, WTg, bg, (void*)gbf, 0);

  // 3) sigma MLP (writes both scratch copy and the sigma output)
  sigma_kernel<<<dim3(ROWS), 256, 0, stream>>>(x, Wu1, bu1, Wu2, bu2, sig, out_sigma);

  // 4) zero the gated-context accumulator (deterministic every call)
  zero_f32_kernel<<<dim3((BB * HH) / 256), 256, 0, stream>>>(Zs, BB * HH);

  // 5) attention: scores (WMMA) -> softmax -> attn out + ctx (WMMA, 8 waves) + gate
  uwca_attn_kernel<<<dim3(MM / 16, NH, BB), 256, 0, stream>>>(qbf, kbf, vT, gbf, sig,
                                                              out_attn, Zs);

  // 6) z = mean @ Wo + bo
  finalize_kernel<<<dim3(BB), 1024, 0, stream>>>(Zs, Wo, bo, out_z);
}
